// SelfAttention_50869592653915
// MI455X (gfx1250) — matrix operand order repaired and checked by reading, not compiled, not run on hardware
//
#include <hip/hip_runtime.h>
#include <math.h>

// ---------------------------------------------------------------------------
// SAGAN self-attention block on gfx1250 (CDNA5), bf16 WMMA pipeline.
//   x:[8,256,64,64] f32, Wf/Wg:[32,256], Wh:[128,256], Wo:[256,128], gamma
// Stage 1: proj GEMMs  -> fT[b][n][32], gfullT[b][n][32], hfullT[b][n][128] (bf16)
//          x tiles staged to LDS with double-buffered GLOBAL_LOAD_ASYNC_TO_LDS
//          (ASYNCcnt-pipelined) when the toolchain exposes the builtin.
// Stage 2: 2x2 maxpool -> gT[b][m][32] (m-major), hP[b][c][m] (c-major)
// Stage 3: fused flash-attention (scores = g^T f so softmax probs land
//          directly in WMMA-A layout) + Wo conv + residual.
// ---------------------------------------------------------------------------

typedef __attribute__((ext_vector_type(16))) __bf16 v16bf;
typedef __attribute__((ext_vector_type(8)))  __bf16 v8bf;
typedef __attribute__((ext_vector_type(8)))  float  v8f;
typedef __attribute__((ext_vector_type(4)))  float  v4f;

#define NB 8
#define CCH 256
#define NPOS 4096   // 64*64
#define MPOS 1024   // 32*32

#ifndef USE_ASYNC_LDS
#define USE_ASYNC_LDS 1
#endif
#if USE_ASYNC_LDS && __has_builtin(__builtin_amdgcn_global_load_async_to_lds_b128)
#define HAVE_ASYNC 1
#else
#define HAVE_ASYNC 0
#endif

static __device__ __forceinline__ v8f wmma_bf16(v16bf a, v16bf b, v8f c) {
  // D = A(16x32) * B(32x16) + C, f32 accumulate
  return __builtin_amdgcn_wmma_f32_16x16x32_bf16(false, a, false, b, (short)0, c,
                                                 false, false);
}

#if HAVE_ASYNC
// The builtin's pointer params are b128 payload pointers: AS1 (global) source
// and AS3 (LDS) destination; payload type is a 16-byte int vector.
typedef int v4i_vs __attribute__((vector_size(16)));
typedef __attribute__((address_space(1))) v4i_vs g_v4i;
typedef __attribute__((address_space(3))) v4i_vs l_v4i;

// Copy one 64-byte x row (16 f32) global -> LDS via the async copy engine.
static __device__ __forceinline__ void async_stage_row(const float* grow,
                                                       float* lrow) {
  g_v4i* g = (g_v4i*)grow;
  l_v4i* l = (l_v4i*)lrow;
  __builtin_amdgcn_global_load_async_to_lds_b128(g, l, 0, 0);
  __builtin_amdgcn_global_load_async_to_lds_b128(g, l, 16, 0);
  __builtin_amdgcn_global_load_async_to_lds_b128(g, l, 32, 0);
  __builtin_amdgcn_global_load_async_to_lds_b128(g, l, 48, 0);
}

static __device__ __forceinline__ void wait_async_le4() {
#if __has_builtin(__builtin_amdgcn_s_wait_asynccnt)
  __builtin_amdgcn_s_wait_asynccnt(4);
#else
  asm volatile("s_wait_asynccnt 0x4" ::: "memory");
#endif
}
static __device__ __forceinline__ void wait_async_le0() {
#if __has_builtin(__builtin_amdgcn_s_wait_asynccnt)
  __builtin_amdgcn_s_wait_asynccnt(0);
#else
  asm volatile("s_wait_asynccnt 0x0" ::: "memory");
#endif
}
#endif  // HAVE_ASYNC

// ---------------------------------------------------------------------------
// Stage 1: Y^T[b][n][o] = sum_c W[o][c] * x[b][c][n]   (one 16x16 tile / wave)
// A = W tile [16o x 32c] (f32->bf16), B = x tile [32c x 16n] staged via LDS.
// ---------------------------------------------------------------------------
__global__ __launch_bounds__(32)
void proj_gemm_kernel(const float* __restrict__ x, const float* __restrict__ W,
                      __bf16* __restrict__ YT, int O) {
  const int lane = threadIdx.x;
  const int col  = lane & 15;
  const int hl   = lane >> 4;
  const int n0 = blockIdx.x * 16;
  const int o0 = blockIdx.y * 16;
  const int b  = blockIdx.z;

  const float* wrow = W + (size_t)(o0 + col) * CCH;
  v8f acc = {};

#if HAVE_ASYNC
  // Double-buffered async staging: f32 x tile [32c x 16n], 17-float row pitch
  // (conflict-free strided reads on the B-build side).
  __shared__ __align__(16) float xs[2][32][17];
  const float* xbase = x + ((size_t)b * CCH + lane) * NPOS + n0;
  async_stage_row(xbase, &xs[0][lane][0]);

  #pragma unroll 2
  for (int kc = 0; kc < CCH / 32; ++kc) {
    const int buf = kc & 1;
    if (kc < CCH / 32 - 1)
      async_stage_row(xbase + (size_t)(kc + 1) * 32 * NPOS, &xs[buf ^ 1][lane][0]);
    if (kc < CCH / 32 - 1) wait_async_le4();  // current buf complete
    else                   wait_async_le0();

    // --- A operand: lane row o=o0+col; elem e -> k=16*hl+e
    union { v16bf v; __bf16 e[16]; } A;
    const float* w0 = wrow + kc * 32 + 16 * hl;
    const float* w1 = w0 + 8;
    #pragma unroll
    for (int j = 0; j < 8; j += 4) {
      v4f t0 = *(const v4f*)(w0 + j);
      v4f t1 = *(const v4f*)(w1 + j);
      #pragma unroll
      for (int q = 0; q < 4; ++q) {
        A.e[j + q]     = (__bf16)t0[q];
        A.e[8 + j + q] = (__bf16)t1[q];
      }
    }
    // --- B operand: lane col n=col; elem e -> c_rel = 16*hl + e
    union { v16bf v; __bf16 e[16]; } Bm;
    #pragma unroll
    for (int e2 = 0; e2 < 16; ++e2)
      Bm.e[e2] = (__bf16)xs[buf][16 * hl + e2][col];
    acc = wmma_bf16(A.v, Bm.v, acc);
  }
#else
  __shared__ __align__(32) __bf16 xT[16][32];   // [n_rel][c_rel]
  #pragma unroll 2
  for (int kc = 0; kc < CCH / 32; ++kc) {
    // --- stage x[kc*32 .. +32][n0 .. +16] into LDS, transposed, as bf16 ---
    const float* xrow = x + ((size_t)b * CCH + kc * 32 + lane) * NPOS + n0;
    float xv[16];
    #pragma unroll
    for (int j = 0; j < 4; ++j) {
      v4f t = *(const v4f*)(xrow + 4 * j);
      xv[4 * j + 0] = t[0]; xv[4 * j + 1] = t[1];
      xv[4 * j + 2] = t[2]; xv[4 * j + 3] = t[3];
    }
    __syncthreads();
    #pragma unroll
    for (int j = 0; j < 16; ++j) xT[j][lane] = (__bf16)xv[j];
    __syncthreads();

    union { v16bf v; __bf16 e[16]; } A;
    const float* w0 = wrow + kc * 32 + 16 * hl;
    const float* w1 = w0 + 8;
    #pragma unroll
    for (int j = 0; j < 8; j += 4) {
      v4f t0 = *(const v4f*)(w0 + j);
      v4f t1 = *(const v4f*)(w1 + j);
      #pragma unroll
      for (int q = 0; q < 4; ++q) {
        A.e[j + q]     = (__bf16)t0[q];
        A.e[8 + j + q] = (__bf16)t1[q];
      }
    }
    v16bf Bm = *(const v16bf*)&xT[col][16 * hl];
    acc = wmma_bf16(A.v, Bm, acc);
  }
#endif

  // D: lane col n=col, vgpr r -> o = o0 + r + 8*hl. Store contiguous run of 8.
  union { v8bf v; __bf16 e[8]; } d;
  #pragma unroll
  for (int r = 0; r < 8; ++r) d.e[r] = (__bf16)acc[r];
  *(v8bf*)(YT + ((size_t)b * NPOS + n0 + col) * O + o0 + 8 * hl) = d.v;
}

// ---------------------------------------------------------------------------
// Stage 2: 2x2 maxpool
// ---------------------------------------------------------------------------
__global__ void pool_g_kernel(const __bf16* __restrict__ gf, __bf16* __restrict__ gT) {
  int tid = blockIdx.x * blockDim.x + threadIdx.x;
  if (tid >= NB * MPOS * 32) return;
  int c8 = tid & 31;
  int m  = (tid >> 5) & (MPOS - 1);
  int b  = tid >> 15;
  int ph = m >> 5, pw = m & 31;
  int n  = (2 * ph) * 64 + 2 * pw;
  const __bf16* base = gf + (size_t)b * NPOS * 32;
  float v0 = (float)base[(size_t)(n)      * 32 + c8];
  float v1 = (float)base[(size_t)(n + 1)  * 32 + c8];
  float v2 = (float)base[(size_t)(n + 64) * 32 + c8];
  float v3 = (float)base[(size_t)(n + 65) * 32 + c8];
  gT[((size_t)b * MPOS + m) * 32 + c8] = (__bf16)fmaxf(fmaxf(v0, v1), fmaxf(v2, v3));
}

__global__ void pool_h_kernel(const __bf16* __restrict__ hf, __bf16* __restrict__ hP) {
  int tid = blockIdx.x * blockDim.x + threadIdx.x;
  if (tid >= NB * 128 * MPOS) return;
  int m = tid & (MPOS - 1);
  int c = (tid >> 10) & 127;
  int b = tid >> 17;
  int ph = m >> 5, pw = m & 31;
  int n  = (2 * ph) * 64 + 2 * pw;
  const __bf16* base = hf + (size_t)b * NPOS * 128;
  float v0 = (float)base[(size_t)(n)      * 128 + c];
  float v1 = (float)base[(size_t)(n + 1)  * 128 + c];
  float v2 = (float)base[(size_t)(n + 64) * 128 + c];
  float v3 = (float)base[(size_t)(n + 65) * 128 + c];
  hP[((size_t)b * 128 + c) * MPOS + m] = (__bf16)fmaxf(fmaxf(v0, v1), fmaxf(v2, v3));
}

// ---------------------------------------------------------------------------
// Stage 3: fused flash attention + Wo conv + residual.
// One wave handles 16 query positions (n-tile).
// Score GEMM computed transposed: St[m,n] = gT[m,:] . fT[n,:]  so the softmax'd
// probabilities are already in WMMA A-operand layout for the o-GEMM.
// ---------------------------------------------------------------------------
__global__ __launch_bounds__(32)
void attn_fused_kernel(const float* __restrict__ x, const float* __restrict__ Wo,
                       const float* __restrict__ gamma_p,
                       const __bf16* __restrict__ fT, const __bf16* __restrict__ gT,
                       const __bf16* __restrict__ hP, float* __restrict__ out) {
  __shared__ __align__(32) __bf16 oT[16][128];  // [n_rel][ci]
  const int lane = threadIdx.x;
  const int col  = lane & 15;
  const int hl   = lane >> 4;
  const int n0 = blockIdx.x * 16;
  const int b  = blockIdx.y;

  // B operand of score GEMM, loop invariant: lane col n; elem e -> c = 16*hl+e.
  v16bf Bf = *(const v16bf*)(fT + ((size_t)b * NPOS + n0 + col) * 32 + 16 * hl);

  v8f acc[8];
  #pragma unroll
  for (int i = 0; i < 8; ++i) acc[i] = (v8f){};
  float run_max = -__builtin_inff();
  float run_sum = 0.0f;

  for (int m0 = 0; m0 < MPOS; m0 += 32) {
    // A operands: gT rows m0+col and m0+16+col; elem e -> c=16*hl+e
    const __bf16* g0 = gT + ((size_t)b * MPOS + m0 + col) * 32;
    const __bf16* g1 = g0 + 16 * 32;
    __builtin_prefetch(g0 + 32 * 32, 0, 0);   // next chunk -> global_prefetch
    union { v16bf v; v8bf p[2]; } A0, A1;
    A0.p[0] = *(const v8bf*)(g0 + 16 * hl);
    A0.p[1] = *(const v8bf*)(g0 + 16 * hl + 8);
    A1.p[0] = *(const v8bf*)(g1 + 16 * hl);
    A1.p[1] = *(const v8bf*)(g1 + 16 * hl + 8);

    v8f z = {};
    v8f S0 = wmma_bf16(A0.v, Bf, z);   // St rows m0 + r + 8*hl,  col n0+col
    v8f S1 = wmma_bf16(A1.v, Bf, z);   // St rows m0+16 + r + 8*hl

    // ---- online softmax over m (this lane: 16 of the 32 m's; partner lane^16 rest)
    float cm = -__builtin_inff();
    #pragma unroll
    for (int r = 0; r < 8; ++r) cm = fmaxf(cm, fmaxf(S0[r], S1[r]));
    cm = fmaxf(cm, __shfl_xor(cm, 16, 32));
    float nm   = fmaxf(run_max, cm);
    float corr = __expf(run_max - nm);
    float p0[8], p1[8], ls = 0.0f;
    #pragma unroll
    for (int r = 0; r < 8; ++r) {
      p0[r] = __expf(S0[r] - nm);
      p1[r] = __expf(S1[r] - nm);
      ls += p0[r] + p1[r];
    }
    ls += __shfl_xor(ls, 16, 32);
    run_sum = run_sum * corr + ls;
    run_max = nm;

    // rescale accumulators: row n = r + 8*hl -> fetch that row's corr factor
    float fr[8];
    #pragma unroll
    for (int r = 0; r < 8; ++r) fr[r] = __shfl(corr, r + 8 * hl, 32);
    #pragma unroll
    for (int ct = 0; ct < 8; ++ct)
      #pragma unroll
      for (int r = 0; r < 8; ++r) acc[ct][r] *= fr[r];

    // probabilities are exactly the A layout of the o-GEMM (no transpose!)
    union { v16bf v; __bf16 e[16]; } P;
    #pragma unroll
    for (int r = 0; r < 8; ++r) {
      P.e[r]     = (__bf16)p0[r];
      P.e[8 + r] = (__bf16)p1[r];
    }

    // o[n, c] += p[n, m] * h[c, m];  B elem e<8 -> m = m0 + 8*hl + e, e>=8 -> m0 + 16 + 8*hl + (e-8) (c-major h)
    #pragma unroll
    for (int ct = 0; ct < 8; ++ct) {
      const __bf16* hrow =
          hP + ((size_t)b * 128 + ct * 16 + col) * MPOS + m0 + 8 * hl;
      union { v16bf v; v8bf p[2]; } Bh;
      Bh.p[0] = *(const v8bf*)hrow;
      Bh.p[1] = *(const v8bf*)(hrow + 16);
      acc[ct] = wmma_bf16(P.v, Bh.v, acc[ct]);
    }
  }

  // ---- normalize and stage o tile [16n x 128c] to LDS (bf16) for Wo conv
  float inv = 1.0f / run_sum;
  float fr[8];
  #pragma unroll
  for (int r = 0; r < 8; ++r) fr[r] = __shfl(inv, r + 8 * hl, 32);
  #pragma unroll
  for (int ct = 0; ct < 8; ++ct)
    #pragma unroll
    for (int r = 0; r < 8; ++r)
      oT[r + 8 * hl][ct * 16 + col] = (__bf16)(acc[ct][r] * fr[r]);
  __syncthreads();

  // ---- out[co, n] = gamma * sum_ci Wo[co, ci] * o[n, ci] + x[co, n]
  const float gamma = *gamma_p;
  #pragma unroll 4
  for (int cot = 0; cot < 16; ++cot) {
    v8f aF = {};
    #pragma unroll
    for (int kc = 0; kc < 4; ++kc) {
      union { v16bf v; __bf16 e[16]; } Aw;
      const float* w0 = Wo + (size_t)(cot * 16 + col) * 128 + kc * 32 + 16 * hl;
      const float* w1 = w0 + 8;
      #pragma unroll
      for (int j = 0; j < 8; j += 4) {
        v4f t0 = *(const v4f*)(w0 + j);
        v4f t1 = *(const v4f*)(w1 + j);
        #pragma unroll
        for (int q = 0; q < 4; ++q) {
          Aw.e[j + q]     = (__bf16)t0[q];
          Aw.e[8 + j + q] = (__bf16)t1[q];
        }
      }
      v16bf Bo = *(const v16bf*)&oT[col][kc * 32 + 16 * hl];
      aF = wmma_bf16(Aw.v, Bo, aF);
    }
    #pragma unroll
    for (int r = 0; r < 8; ++r) {
      int co = cot * 16 + r + 8 * hl;
      size_t idx = ((size_t)b * CCH + co) * NPOS + n0 + col;
      out[idx] = gamma * aF[r] + x[idx];
    }
  }
}

// ---------------------------------------------------------------------------
extern "C" void kernel_launch(void* const* d_in, const int* in_sizes, int n_in,
                              void* d_out, int out_size, void* d_ws, size_t ws_size,
                              hipStream_t stream) {
  const float* x     = (const float*)d_in[0];
  const float* Wf    = (const float*)d_in[1];
  const float* Wg    = (const float*)d_in[2];
  const float* Wh    = (const float*)d_in[3];
  const float* Wo    = (const float*)d_in[4];
  const float* gamma = (const float*)d_in[5];

  char* ws = (char*)d_ws;
  __bf16* fT  = (__bf16*)(ws);                        // [8][4096][32]  2 MB
  __bf16* gfT = (__bf16*)(ws + ((size_t)2 << 20));    // [8][4096][32]  2 MB
  __bf16* hfT = (__bf16*)(ws + ((size_t)4 << 20));    // [8][4096][128] 8 MB
  __bf16* gT  = (__bf16*)(ws + ((size_t)12 << 20));   // [8][1024][32]  0.5 MB
  __bf16* hP  = (__bf16*)(ws + ((size_t)13 << 20));   // [8][128][1024] 2 MB

  dim3 wave(32);
  proj_gemm_kernel<<<dim3(NPOS / 16, 32 / 16, NB), wave, 0, stream>>>(x, Wf, fT, 32);
  proj_gemm_kernel<<<dim3(NPOS / 16, 32 / 16, NB), wave, 0, stream>>>(x, Wg, gfT, 32);
  proj_gemm_kernel<<<dim3(NPOS / 16, 128 / 16, NB), wave, 0, stream>>>(x, Wh, hfT, 128);

  pool_g_kernel<<<(NB * MPOS * 32) / 256, 256, 0, stream>>>(gfT, gT);
  pool_h_kernel<<<(NB * 128 * MPOS) / 256, 256, 0, stream>>>(hfT, hP);

  attn_fused_kernel<<<dim3(NPOS / 16, NB), wave, 0, stream>>>(
      x, Wo, gamma, fT, gT, hP, (float*)d_out);
}
